// SinkhornTransformerEncoder_51144470560900
// MI455X (gfx1250) — compile-verified, matrix operand order checked by reading, not run on hardware
//
#include <hip/hip_runtime.h>

typedef _Float16 half_t;
typedef __attribute__((ext_vector_type(16))) _Float16 v16h;
typedef __attribute__((ext_vector_type(8)))  _Float16 v8h;
typedef __attribute__((ext_vector_type(8)))  float    v8f;

#define CB   4
#define CS   2048
#define CD   512
#define CH   8
#define CDH  64
#define CL   4
#define CMLP 2048
#define CNB  16
#define CBLK 128
#define CBS  (CB*CS)
#define NEGBIG (-1e9f)

// ---------------------------------------------------------------------------
// Fragment gather matching CDNA5 WMMA 16-bit layouts (CDNA5 ISA):
// lane<16 holds K elements {0..7,16..23}; lane>=16 holds {8..15,24..31}.
// Caller passes per-lane base pointer p (row/col already selected, +hi*8):
// elements are p[0..7] and p[16..23].
// ---------------------------------------------------------------------------
__device__ inline v16h load_frag_rowmaj(const half_t* p) {
    v8h lo = *(const v8h*)p;
    v8h hi = *(const v8h*)(p + 16);
    v16h r;
#pragma unroll
    for (int e = 0; e < 8; ++e) { r[e] = lo[e]; r[e + 8] = hi[e]; }
    return r;
}

// CDNA5 async direct-to-LDS copy (16B per lane), tracked by ASYNCcnt.
__device__ inline void async_g2l_b128(unsigned lds_addr, const half_t* gaddr) {
    unsigned long long ga = (unsigned long long)(uintptr_t)gaddr;
    asm volatile("global_load_async_to_lds_b128 %0, %1, off"
                 :: "v"(lds_addr), "v"(ga) : "memory");
}
__device__ inline void wait_asynccnt0() {
    asm volatile("s_wait_asynccnt 0x0" ::: "memory");
}

// ---------------------------------------------------------------------------
// fp32 -> fp16 conversion
// ---------------------------------------------------------------------------
__global__ void cvt_f16_kernel(const float* __restrict__ src, half_t* __restrict__ dst, int n) {
    int i = blockIdx.x * blockDim.x + threadIdx.x;
    if (i < n) dst[i] = (half_t)src[i];
}

// ---------------------------------------------------------------------------
// x = embed[token] + sinusoidal PE ; also key-padding mask mb
// ---------------------------------------------------------------------------
__global__ void embed_kernel(const int* __restrict__ tok, const float* __restrict__ emb,
                             float* __restrict__ X, float* __restrict__ mb) {
    int idx = blockIdx.x * blockDim.x + threadIdx.x;
    if (idx >= CBS * CD) return;
    int d  = idx & (CD - 1);
    int bs = idx >> 9;
    int s  = bs & (CS - 1);
    int t  = tok[bs];
    int i2 = (d >> 1) * 2;
    float freq = expf(-(float)i2 * (logf(10000.0f) / (float)CD));
    float ang  = (float)s * freq;
    float pe   = (d & 1) ? cosf(ang) : sinf(ang);
    X[idx] = emb[(size_t)t * CD + d] + pe;
    if (d == 0) mb[bs] = (t > 0) ? 1.0f : 0.0f;
}

// ---------------------------------------------------------------------------
// LayerNorm over D=512, one 256-thread block per row. outh!=null -> fp16 out,
// else fp32 out.
// ---------------------------------------------------------------------------
__global__ __launch_bounds__(256) void ln_kernel(const float* __restrict__ x,
                                                 const float* __restrict__ sc,
                                                 const float* __restrict__ bi,
                                                 half_t* outh, float* outf) {
    int row = blockIdx.x, t = threadIdx.x;
    const float* xr = x + (size_t)row * CD;
    float a = xr[t], c = xr[t + 256];
    __shared__ float red[256];
    red[t] = a + c; __syncthreads();
    for (int s = 128; s > 0; s >>= 1) { if (t < s) red[t] += red[t + s]; __syncthreads(); }
    float mean = red[0] * (1.0f / CD);
    __syncthreads();
    float da = a - mean, dc = c - mean;
    red[t] = da * da + dc * dc; __syncthreads();
    for (int s = 128; s > 0; s >>= 1) { if (t < s) red[t] += red[t + s]; __syncthreads(); }
    float rs = rsqrtf(red[0] * (1.0f / CD) + 1e-6f);
    float o0 = da * rs * sc[t]       + bi[t];
    float o1 = dc * rs * sc[t + 256] + bi[t + 256];
    size_t o = (size_t)row * CD;
    if (outh) { outh[o + t] = (half_t)o0; outh[o + t + 256] = (half_t)o1; }
    else      { outf[o + t] = o0;         outf[o + t + 256] = o1; }
}

// ---------------------------------------------------------------------------
// Generic WMMA GEMM: C[M,N] = A[M,K] x W[K,N], fp16 inputs, fp32 accum.
// mode 0: outH = acc*scale ; mode 1: outH = gelu(acc+bias) ;
// mode 2: outF += acc + (bias?bias[n]:0)
// Block 256 = 8 waves; 128x64 tile; wave tile 32x32; K-step 32.
// Double-buffered LDS; A staged via async global->LDS (ASYNCcnt), B staged
// through VGPRs (needs transpose) with its wait landing after the WMMAs.
// ---------------------------------------------------------------------------
__global__ __launch_bounds__(256) void gemm_wmma_kernel(
    const half_t* __restrict__ A, const half_t* __restrict__ W,
    int M, int K, int N, int mode, float scale,
    const float* __restrict__ bias, half_t* outH, float* outF) {
    __shared__ half_t As[2][128][40];
    __shared__ half_t Bs[2][64][40];   // transposed: Bs[.][n][k]
    int t = threadIdx.x;
    int wv = t >> 5, lane = t & 31, lo = lane & 15, hi = lane >> 4;
    int wm = wv & 3, wn = wv >> 2;
    int m0 = blockIdx.x * 128, n0 = blockIdx.y * 64;

    // staging coordinates
    int arow = t >> 2;          // 0..63 (rows arow and arow+64)
    int akp  = (t & 3) * 8;     // 0,8,16,24
    int bkr  = t >> 3;          // 0..31
    int bnc  = (t & 7) * 8;     // 0..56

    const half_t* Ag0 = A + (size_t)(m0 + arow) * K + akp;
    const half_t* Ag1 = A + (size_t)(m0 + arow + 64) * K + akp;
    const half_t* Wg  = W + (size_t)bkr * N + n0 + bnc;

    int nk = K >> 5;
    v8f acc[2][2] = {};

    // prologue: stage tile 0 into buffer 0
    async_g2l_b128((unsigned)(uintptr_t)&As[0][arow][akp],      Ag0);
    async_g2l_b128((unsigned)(uintptr_t)&As[0][arow + 64][akp], Ag1);
    {
        v8h bv = *(const v8h*)Wg;
#pragma unroll
        for (int j = 0; j < 8; ++j) Bs[0][bnc + j][bkr] = bv[j];
    }
    wait_asynccnt0();
    __syncthreads();

    for (int ki = 0; ki < nk; ++ki) {
        int cur = ki & 1;
        bool hasNext = (ki + 1) < nk;
        v8h bv;
        if (hasNext) {
            // issue next tile's loads before this tile's compute
            async_g2l_b128((unsigned)(uintptr_t)&As[cur ^ 1][arow][akp],
                           Ag0 + (size_t)(ki + 1) * 32);
            async_g2l_b128((unsigned)(uintptr_t)&As[cur ^ 1][arow + 64][akp],
                           Ag1 + (size_t)(ki + 1) * 32);
            bv = *(const v8h*)(Wg + (size_t)(ki + 1) * 32 * N);
        }
        v16h afr[2], bfr[2];
        afr[0] = load_frag_rowmaj(&As[cur][wm * 32 + lo][hi * 8]);
        afr[1] = load_frag_rowmaj(&As[cur][wm * 32 + 16 + lo][hi * 8]);
        bfr[0] = load_frag_rowmaj(&Bs[cur][wn * 32 + lo][hi * 8]);
        bfr[1] = load_frag_rowmaj(&Bs[cur][wn * 32 + 16 + lo][hi * 8]);
#pragma unroll
        for (int ti = 0; ti < 2; ++ti)
#pragma unroll
            for (int tj = 0; tj < 2; ++tj)
                acc[ti][tj] = __builtin_amdgcn_wmma_f32_16x16x32_f16(
                    false, afr[ti], false, bfr[tj], (short)0, acc[ti][tj], false, false);
        if (hasNext) {
#pragma unroll
            for (int j = 0; j < 8; ++j) Bs[cur ^ 1][bnc + j][bkr] = bv[j];
            wait_asynccnt0();
            __syncthreads();
        }
    }
#pragma unroll
    for (int ti = 0; ti < 2; ++ti)
#pragma unroll
        for (int tj = 0; tj < 2; ++tj)
#pragma unroll
            for (int r = 0; r < 8; ++r) {
                int m = m0 + wm * 32 + ti * 16 + r + hi * 8;
                int n = n0 + wn * 32 + tj * 16 + lo;
                float v = acc[ti][tj][r];
                size_t o = (size_t)m * N + n;
                if (mode == 0) {
                    outH[o] = (half_t)(v * scale);
                } else if (mode == 1) {
                    float x = v + bias[n];
                    float g = 0.5f * x * (1.0f + tanhf(0.7978845608f * (x + 0.044715f * x * x * x)));
                    outH[o] = (half_t)g;
                } else {
                    float x = v + (bias ? bias[n] : 0.0f);
                    outF[o] += x;
                }
            }
}

// ---------------------------------------------------------------------------
// ksum[b,n,h,f] = sum_c k[b, n*128+c, h, f]
// ---------------------------------------------------------------------------
__global__ void ksum_kernel(const half_t* __restrict__ k16, float* __restrict__ ksum) {
    int idx = blockIdx.x * blockDim.x + threadIdx.x;
    if (idx >= CB * CNB * CH * CDH) return;
    int f = idx & 63, h = (idx >> 6) & 7, n = (idx >> 9) & 15, b = idx >> 13;
    const half_t* base = k16 + ((size_t)(b * CS + n * CBLK)) * CD + h * CDH + f;
    float s = 0.0f;
    for (int c = 0; c < CBLK; ++c) s += (float)base[(size_t)c * CD];
    ksum[idx] = s;
}

// ---------------------------------------------------------------------------
// logits[b,h,n,m] = sum_f ksum[b,n,h,f] * sort_w[h,f,m]
// ---------------------------------------------------------------------------
__global__ void logits_kernel(const float* __restrict__ ksum, const float* __restrict__ sw,
                              float* __restrict__ logi) {
    int idx = blockIdx.x * blockDim.x + threadIdx.x;
    if (idx >= CB * CH * CNB * CNB) return;
    int m = idx & 15, n = (idx >> 4) & 15, h = (idx >> 8) & 7, b = idx >> 11;
    const float* kp = ksum + ((size_t)((b * CNB + n) * CH + h)) * CDH;
    float s = 0.0f;
    for (int f = 0; f < CDH; ++f) s += kp[f] * sw[(h * CDH + f) * CNB + m];
    logi[idx] = s;
}

// ---------------------------------------------------------------------------
// Sinkhorn: 8 iterations of row/col log-normalization on 16x16, one block per (b,h)
// ---------------------------------------------------------------------------
__global__ __launch_bounds__(256) void sinkhorn_kernel(const float* __restrict__ logi,
                                                       float* __restrict__ perm) {
    int mat = blockIdx.x;
    __shared__ float la[16][17];
    __shared__ float lse[16];
    int t = threadIdx.x, m = t & 15, n = t >> 4;
    la[n][m] = logi[mat * 256 + n * 16 + m];
    __syncthreads();
    for (int it = 0; it < 8; ++it) {
        if (t < 16) {
            float mx = -3.4e38f;
            for (int j = 0; j < 16; ++j) mx = fmaxf(mx, la[t][j]);
            float s = 0.0f;
            for (int j = 0; j < 16; ++j) s += expf(la[t][j] - mx);
            lse[t] = logf(s) + mx;
        }
        __syncthreads();
        la[n][m] -= lse[n];
        __syncthreads();
        if (t < 16) {
            float mx = -3.4e38f;
            for (int j = 0; j < 16; ++j) mx = fmaxf(mx, la[j][t]);
            float s = 0.0f;
            for (int j = 0; j < 16; ++j) s += expf(la[j][t] - mx);
            lse[t] = logf(s) + mx;
        }
        __syncthreads();
        la[n][m] -= lse[m];
        __syncthreads();
    }
    perm[mat * 256 + n * 16 + m] = expf(la[n][m]);
}

// ---------------------------------------------------------------------------
// sk/sv[b,n,c,h,f] = sum_m perm[b,h,n,m] * k/v[b, m*128+c, h, f]
// ---------------------------------------------------------------------------
__global__ void skv_kernel(const half_t* __restrict__ k16, const half_t* __restrict__ v16,
                           const float* __restrict__ perm,
                           half_t* __restrict__ sk, half_t* __restrict__ sv) {
    int idx = blockIdx.x * blockDim.x + threadIdx.x;
    if (idx >= CB * CNB * CBLK * CD) return;
    int col = idx & 511, c = (idx >> 9) & 127, n = (idx >> 16) & 15, b = idx >> 20;
    int h = col >> 6;
    const float* pr = perm + ((b * CH + h) * CNB + n) * CNB;
    size_t base = ((size_t)(b * CS) + c) * CD + col;
    float aK = 0.0f, aV = 0.0f;
    for (int m = 0; m < CNB; ++m) {
        float w = pr[m];
        size_t o = base + (size_t)m * CBLK * CD;
        aK += w * (float)k16[o];
        aV += w * (float)v16[o];
    }
    sk[idx] = (half_t)aK;
    sv[idx] = (half_t)aV;
}

// ---------------------------------------------------------------------------
// fb[b,h,n,k] (k in 0..255): 0 or -1e9 additive mask bias
// ---------------------------------------------------------------------------
__global__ void fbias_kernel(const float* __restrict__ mb, const float* __restrict__ perm,
                             float* __restrict__ fb) {
    int idx = blockIdx.x * blockDim.x + threadIdx.x;
    if (idx >= CB * CH * CNB * 256) return;
    int k = idx & 255, n = (idx >> 8) & 15, h = (idx >> 12) & 7, b = idx >> 15;
    float bias;
    if (k < CBLK) {
        bias = (mb[b * CS + n * CBLK + k] > 0.5f) ? 0.0f : NEGBIG;
    } else {
        int c = k - CBLK;
        const float* pr = perm + ((b * CH + h) * CNB + n) * CNB;
        float s = 0.0f;
        for (int m = 0; m < CNB; ++m) s += pr[m] * mb[b * CS + m * CBLK + c];
        bias = (s > 0.5f) ? 0.0f : NEGBIG;
    }
    fb[idx] = bias;
}

// ---------------------------------------------------------------------------
// vT[b,n,h,f,k] = vcat[b,n,k,h,f]  (k in 0..255; k<128 local, else sorted)
// ---------------------------------------------------------------------------
__global__ void vtrans_kernel(const half_t* __restrict__ v16, const half_t* __restrict__ sv,
                              half_t* __restrict__ vT) {
    int idx = blockIdx.x * blockDim.x + threadIdx.x;
    if (idx >= CB * CNB * CH * CDH * 256) return;
    int k = idx & 255, f = (idx >> 8) & 63, h = (idx >> 14) & 7, n = (idx >> 17) & 15, b = idx >> 21;
    half_t val;
    if (k < CBLK)
        val = v16[((size_t)(b * CS + n * CBLK + k)) * CD + h * CDH + f];
    else
        val = sv[((size_t)((b * CNB + n) * CBLK + (k - CBLK))) * CD + h * CDH + f];
    vT[idx] = val;
}

// ---------------------------------------------------------------------------
// Block attention: one 2-wave workgroup per (b,h,block,q-quarter); each wave
// owns 16 q rows. Scores via WMMA, masked softmax in LDS, P x V via WMMA.
// ---------------------------------------------------------------------------
__global__ __launch_bounds__(64) void attn_kernel(
    const half_t* __restrict__ q16, const half_t* __restrict__ k16,
    const half_t* __restrict__ sk, const half_t* __restrict__ vT,
    const float* __restrict__ fb, half_t* __restrict__ o16) {
    __shared__ float  Ssc[2][16][264];
    __shared__ half_t Qs[2][16][72];
    __shared__ float  maskb[256];
    int g = blockIdx.x;
    int qg = g & 3, n = (g >> 2) & 15, h = (g >> 6) & 7, b = g >> 9;
    int t = threadIdx.x, wv = t >> 5, lane = t & 31, lo = lane & 15, hi = lane >> 4;

    for (int k = t; k < 256; k += 64)
        maskb[k] = fb[(((b * CH + h) * CNB + n) << 8) + k];
    for (int i = t; i < 2 * 16 * CDH; i += 64) {
        int w = i >> 10, rem = i & 1023, r = rem >> 6, c = rem & 63;
        int srow = n * CBLK + qg * 32 + w * 16 + r;
        Qs[w][r][c] = q16[((size_t)(b * CS + srow)) * CD + h * CDH + c];
    }
    __syncthreads();

    // scores 16x256
    for (int kt = 0; kt < 16; ++kt) {
        int key = kt * 16 + lo;
        const half_t* krow = (key < CBLK)
            ? k16 + ((size_t)(b * CS + n * CBLK + key)) * CD + h * CDH
            : sk  + ((size_t)((b * CNB + n) * CBLK + (key - CBLK))) * CD + h * CDH;
        v8f acc = {};
#pragma unroll
        for (int kc = 0; kc < 2; ++kc) {
            v16h a  = load_frag_rowmaj(&Qs[wv][lo][kc * 32 + hi * 8]);
            v16h bf = load_frag_rowmaj(krow + kc * 32 + hi * 8);
            acc = __builtin_amdgcn_wmma_f32_16x16x32_f16(false, a, false, bf, (short)0, acc, false, false);
        }
#pragma unroll
        for (int r = 0; r < 8; ++r) Ssc[wv][r + hi * 8][kt * 16 + lo] = acc[r];
    }
    __syncthreads();

    // masked softmax per row (lanes 0..15 each own one row)
    if (lane < 16) {
        float* row = &Ssc[wv][lane][0];
        float mx = -3.4e38f;
        for (int k = 0; k < 256; ++k) { float s = row[k] + maskb[k]; row[k] = s; mx = fmaxf(mx, s); }
        float sum = 0.0f;
        for (int k = 0; k < 256; ++k) { float e = __expf(row[k] - mx); row[k] = e; sum += e; }
        float inv = 1.0f / sum;
        for (int k = 0; k < 256; ++k) row[k] *= inv;
    }
    __syncthreads();

    // O = P x Vcat  (16x256 x 256x64)
    v8f oacc[4] = {};
    for (int kc = 0; kc < 8; ++kc) {
        const float* pp = &Ssc[wv][lo][kc * 32 + hi * 8];
        v16h a;
#pragma unroll
        for (int e = 0; e < 8; ++e) { a[e] = (half_t)pp[e]; a[e + 8] = (half_t)pp[16 + e]; }
#pragma unroll
        for (int ft = 0; ft < 4; ++ft) {
            const half_t* vp = vT + ((((size_t)((b * CNB + n) * CH + h)) * CDH + ft * 16 + lo) * 256)
                                  + kc * 32 + hi * 8;
            v16h bf = load_frag_rowmaj(vp);
            oacc[ft] = __builtin_amdgcn_wmma_f32_16x16x32_f16(false, a, false, bf, (short)0, oacc[ft], false, false);
        }
    }
    int qbase = qg * 32 + wv * 16;
#pragma unroll
    for (int ft = 0; ft < 4; ++ft)
#pragma unroll
        for (int r = 0; r < 8; ++r) {
            int srow = n * CBLK + qbase + r + hi * 8;
            o16[((size_t)(b * CS + srow)) * CD + h * CDH + ft * 16 + lo] = (half_t)oacc[ft][r];
        }
}

// ---------------------------------------------------------------------------
// Workspace layout (bytes, all 256-aligned)
// ---------------------------------------------------------------------------
#define OFF_X    ((size_t)0)
#define OFF_H16  ((size_t)16777216)
#define OFF_Q16  ((size_t)25165824)
#define OFF_K16  ((size_t)33554432)
#define OFF_V16  ((size_t)41943040)
#define OFF_O16  ((size_t)50331648)
#define OFF_SK   ((size_t)58720256)
#define OFF_SV   ((size_t)67108864)
#define OFF_VT   ((size_t)75497472)
#define OFF_Y16  ((size_t)92274688)
#define OFF_KSUM ((size_t)125829120)
#define OFF_LOGI ((size_t)125960192)
#define OFF_PERM ((size_t)125992960)
#define OFF_MB   ((size_t)126025728)
#define OFF_FB   ((size_t)126058496)
#define OFF_WQ16 ((size_t)126582784)
#define OFF_WK16 ((size_t)128679936)
#define OFF_WV16 ((size_t)130777088)
#define OFF_WO16 ((size_t)132874240)
#define OFF_W116 ((size_t)134971392)
#define OFF_W216 ((size_t)143360000)

extern "C" void kernel_launch(void* const* d_in, const int* in_sizes, int n_in,
                              void* d_out, int out_size, void* d_ws, size_t ws_size,
                              hipStream_t stream) {
    const int*   tokens = (const int*)  d_in[0];
    const float* embedw = (const float*)d_in[1];
    const float* ln1_s  = (const float*)d_in[2];
    const float* ln1_b  = (const float*)d_in[3];
    const float* wq     = (const float*)d_in[4];
    const float* wk     = (const float*)d_in[5];
    const float* wvp    = (const float*)d_in[6];
    const float* wo     = (const float*)d_in[7];
    const float* sortw  = (const float*)d_in[8];
    const float* ln2_s  = (const float*)d_in[9];
    const float* ln2_b  = (const float*)d_in[10];
    const float* w1     = (const float*)d_in[11];
    const float* b1     = (const float*)d_in[12];
    const float* w2     = (const float*)d_in[13];
    const float* b2     = (const float*)d_in[14];
    const float* lnf_s  = (const float*)d_in[15];
    const float* lnf_b  = (const float*)d_in[16];
    float* out = (float*)d_out;

    char* ws = (char*)d_ws;
    float*  X    = (float*) (ws + OFF_X);
    half_t* H16  = (half_t*)(ws + OFF_H16);
    half_t* Q16  = (half_t*)(ws + OFF_Q16);
    half_t* K16  = (half_t*)(ws + OFF_K16);
    half_t* V16  = (half_t*)(ws + OFF_V16);
    half_t* O16  = (half_t*)(ws + OFF_O16);
    half_t* SK   = (half_t*)(ws + OFF_SK);
    half_t* SV   = (half_t*)(ws + OFF_SV);
    half_t* VT   = (half_t*)(ws + OFF_VT);
    half_t* Y16  = (half_t*)(ws + OFF_Y16);
    float*  KSUM = (float*) (ws + OFF_KSUM);
    float*  LOGI = (float*) (ws + OFF_LOGI);
    float*  PERM = (float*) (ws + OFF_PERM);
    float*  MB   = (float*) (ws + OFF_MB);
    float*  FB   = (float*) (ws + OFF_FB);
    half_t* WQ16 = (half_t*)(ws + OFF_WQ16);
    half_t* WK16 = (half_t*)(ws + OFF_WK16);
    half_t* WV16 = (half_t*)(ws + OFF_WV16);
    half_t* WO16 = (half_t*)(ws + OFF_WO16);
    half_t* W116 = (half_t*)(ws + OFF_W116);
    half_t* W216 = (half_t*)(ws + OFF_W216);

    const int nDD  = CL * CD * CD;    // 1,048,576
    const int nDM  = CL * CD * CMLP;  // 4,194,304
    cvt_f16_kernel<<<(nDD + 255) / 256, 256, 0, stream>>>(wq,  WQ16, nDD);
    cvt_f16_kernel<<<(nDD + 255) / 256, 256, 0, stream>>>(wk,  WK16, nDD);
    cvt_f16_kernel<<<(nDD + 255) / 256, 256, 0, stream>>>(wvp, WV16, nDD);
    cvt_f16_kernel<<<(nDD + 255) / 256, 256, 0, stream>>>(wo,  WO16, nDD);
    cvt_f16_kernel<<<(nDM + 255) / 256, 256, 0, stream>>>(w1,  W116, nDM);
    cvt_f16_kernel<<<(nDM + 255) / 256, 256, 0, stream>>>(w2,  W216, nDM);

    embed_kernel<<<(CBS * CD + 255) / 256, 256, 0, stream>>>(tokens, embedw, X, MB);

    for (int l = 0; l < CL; ++l) {
        // LN1 -> H16 (fp16)
        ln_kernel<<<CBS, 256, 0, stream>>>(X, ln1_s + l * CD, ln1_b + l * CD, H16, nullptr);
        // Q/K/V projections (Q scaled by 1/sqrt(DH))
        dim3 g1(CBS / 128, CD / 64);
        gemm_wmma_kernel<<<g1, 256, 0, stream>>>(H16, WQ16 + (size_t)l * CD * CD,
            CBS, CD, CD, 0, 0.125f, nullptr, Q16, nullptr);
        gemm_wmma_kernel<<<g1, 256, 0, stream>>>(H16, WK16 + (size_t)l * CD * CD,
            CBS, CD, CD, 0, 1.0f, nullptr, K16, nullptr);
        gemm_wmma_kernel<<<g1, 256, 0, stream>>>(H16, WV16 + (size_t)l * CD * CD,
            CBS, CD, CD, 0, 1.0f, nullptr, V16, nullptr);
        // sort net
        ksum_kernel<<<(CB * CNB * CH * CDH + 255) / 256, 256, 0, stream>>>(K16, KSUM);
        logits_kernel<<<(CB * CH * CNB * CNB + 255) / 256, 256, 0, stream>>>(
            KSUM, sortw + (size_t)l * CH * CDH * CNB, LOGI);
        sinkhorn_kernel<<<CB * CH, 256, 0, stream>>>(LOGI, PERM);
        // sorted key/value blocks + mask bias + transposed Vcat
        skv_kernel<<<(CB * CNB * CBLK * CD + 255) / 256, 256, 0, stream>>>(K16, V16, PERM, SK, SV);
        fbias_kernel<<<(CB * CH * CNB * 256 + 255) / 256, 256, 0, stream>>>(MB, PERM, FB);
        vtrans_kernel<<<(CB * CNB * CH * CDH * 256 + 255) / 256, 256, 0, stream>>>(V16, SV, VT);
        // attention
        attn_kernel<<<CB * CH * CNB * 4, 64, 0, stream>>>(Q16, K16, SK, VT, FB, O16);
        // output projection, residual add into X
        gemm_wmma_kernel<<<g1, 256, 0, stream>>>(O16, WO16 + (size_t)l * CD * CD,
            CBS, CD, CD, 2, 1.0f, nullptr, nullptr, X);
        // MLP
        ln_kernel<<<CBS, 256, 0, stream>>>(X, ln2_s + l * CD, ln2_b + l * CD, H16, nullptr);
        dim3 g2(CBS / 128, CMLP / 64);
        gemm_wmma_kernel<<<g2, 256, 0, stream>>>(H16, W116 + (size_t)l * CD * CMLP,
            CBS, CD, CMLP, 1, 1.0f, b1 + l * CMLP, Y16, nullptr);
        dim3 g3(CBS / 128, CD / 64);
        gemm_wmma_kernel<<<g3, 256, 0, stream>>>(Y16, W216 + (size_t)l * CMLP * CD,
            CBS, CMLP, CD, 2, 1.0f, b2 + l * CD, nullptr, X);
    }
    // final LN -> fp32 output
    ln_kernel<<<CBS, 256, 0, stream>>>(X, lnf_s, lnf_b, nullptr, out);
}
